// GAN_25520695673230
// MI455X (gfx1250) — matrix operand order repaired and checked by reading, not compiled, not run on hardware
//
#include <hip/hip_runtime.h>

#define B_ 4
#define N_ 2048
#define F_ 128
#define D_ 128
#define H_ 4

typedef __attribute__((ext_vector_type(16))) __bf16 v16bf;
typedef __attribute__((ext_vector_type(8)))  __bf16 v8bf;
typedef __attribute__((ext_vector_type(8)))  float  v8f;

static __device__ __forceinline__ __bf16 f2bf(float f) {
  unsigned u = __builtin_bit_cast(unsigned, f);
  unsigned r = u + 0x7FFFu + ((u >> 16) & 1u);   // round-to-nearest-even
  unsigned short hbits = (unsigned short)(r >> 16);
  return __builtin_bit_cast(__bf16, hbits);
}

// tanh via v_exp_f32/v_rcp_f32 (~7 VALU ops vs libm's ~15-20). Safe for all
// |x|: exp(-2|x|) underflows to 0 -> result +/-1, no NaN path.
static __device__ __forceinline__ float fast_tanh(float x) {
  const float ax = __builtin_fabsf(x);
  const float t  = __expf(-2.0f * ax);
  const float r  = (1.0f - t) * __builtin_amdgcn_rcpf(1.0f + t);
  return __builtin_copysignf(r, x);
}

static __device__ __forceinline__ v16bf cat8(v8bf lo, v8bf hi) {
  return __builtin_shufflevector(lo, hi, 0,1,2,3,4,5,6,7,8,9,10,11,12,13,14,15);
}

static __device__ __forceinline__ v8f wmma_bf16(v16bf a, v16bf b, v8f c) {
  // D = A(16x32 bf16) x B(32x16 bf16) + C(16x16 f32)
  return __builtin_amdgcn_wmma_f32_16x16x32_bf16(false, a, false, b, (short)0, c,
                                                 false, false);
}

// ---------------------------------------------------------------------------
// Prep: Xp = X @ kernel + bias; store bf16 row-major Xp[b][n][d] and
// transposed XpT[b][d][n] (so it can serve as a contiguous WMMA B operand).
// ---------------------------------------------------------------------------
__global__ void __launch_bounds__(128)
k_prep(const float* __restrict__ X, const float* __restrict__ Kn,
       const float* __restrict__ bias, __bf16* __restrict__ Xp,
       __bf16* __restrict__ XpT) {
  const int row = blockIdx.x;                 // b*N + n
  const int d   = threadIdx.x;                // 0..127
  const float* xr = X + (size_t)row * F_;
  float acc = bias[d];
#pragma unroll 4
  for (int f = 0; f < F_; ++f) acc = fmaf(xr[f], Kn[f * D_ + d], acc);
  const int b = row >> 11;                    // N = 2048
  const int n = row & (N_ - 1);
  const __bf16 v = f2bf(acc);
  Xp[(size_t)row * D_ + d] = v;
  XpT[((size_t)b * D_ + d) * N_ + n] = v;
}

// attnT[h][e][d] = attn[h][d][e]  (bf16) — contiguous B operand for W = Xp@attn
__global__ void __launch_bounds__(256)
k_attnT(const float* __restrict__ attn, __bf16* __restrict__ attnT) {
  const int i = blockIdx.x * 256 + threadIdx.x;  // over H*D*D
  const int h = i >> 14;                         // D*D = 16384
  const int r = i & 16383;
  const int d = r >> 7;
  const int e = r & 127;
  attnT[(((size_t)h * D_) + e) * D_ + d] = f2bf(attn[i]);
}

// ---------------------------------------------------------------------------
// Fused main kernel. Block = (b, 64-row n-tile), 256 threads = 8 waves.
// Wave w: strip = w>>1 (16-row n strip), wcol = w&1 (column half).
// Heads are the INNER loop: all 4 W tiles live in LDS, so each adjacency
// value is loaded from global once per m-chunk and reused by all 4 heads.
// ---------------------------------------------------------------------------
__global__ void __launch_bounds__(256)
k_main(const float* __restrict__ X, const float* __restrict__ Aadj,
       const __bf16* __restrict__ Xp, const __bf16* __restrict__ XpT,
       const __bf16* __restrict__ attnT, float* __restrict__ out) {
  __shared__ __align__(32) __bf16 Ws[H_][64][128];  // 4 W tiles (64 KB)
  __shared__ __align__(32) __bf16 Am[64][64];       // tanh tile  (8 KB)

  const int b    = blockIdx.x >> 5;              // 32 n-tiles per batch
  const int n0   = (blockIdx.x & 31) * 64;
  const int lane = threadIdx.x & 31;
  const int wave = threadIdx.x >> 5;
  const int strip = wave >> 1;                   // 0..3
  const int wcol  = wave & 1;                    // 0..1
  const int lr = lane & 15;
  const int lh = lane >> 4;                      // 0/1

  const __bf16* XpB  = Xp   + (size_t)b * N_ * D_;
  const __bf16* XpTB = XpT  + (size_t)b * D_ * N_;
  const float*  AB   = Aadj + (size_t)b * N_ * N_;

  // ---- Phase 0: Ws[h][n][e] = sum_d Xp[n][d] * attn[h][d][e] (all heads) --
#pragma unroll 1
  for (int h = 0; h < H_; ++h) {
    const __bf16* atT = attnT + (size_t)h * D_ * D_;
    v8f wacc[4] = {};
    const int nrow = n0 + 16 * strip + lr;
    const __bf16* xpr = XpB + (size_t)nrow * D_;
#pragma unroll
    for (int kb = 0; kb < D_; kb += 32) {
      const v16bf a = cat8(*(const v8bf*)(xpr + kb + 16 * lh),
                           *(const v8bf*)(xpr + kb + 16 * lh + 8));
#pragma unroll
      for (int t = 0; t < 4; ++t) {
        const int e = 64 * wcol + 16 * t + lr;
        const v16bf bb = *(const v16bf*)(atT + (size_t)e * D_ + kb + 16 * lh);
        wacc[t] = wmma_bf16(a, bb, wacc[t]);
      }
    }
#pragma unroll
    for (int t = 0; t < 4; ++t)
#pragma unroll
      for (int j = 0; j < 8; ++j)
        Ws[h][16 * strip + j + 8 * lh][64 * wcol + 16 * t + lr] =
            f2bf(wacc[t][j]);
  }
  __syncthreads();

  v8f hacc[4] = {};                              // heads-sum, 64-col slice

#pragma unroll 1
  for (int m0 = 0; m0 < N_; m0 += 64) {
    // Adjacency values for this wave's elementwise footprint: loaded ONCE
    // per m-chunk into registers, reused by all 4 heads.
    float av[2][8];
#pragma unroll
    for (int t = 0; t < 2; ++t)
#pragma unroll
      for (int j = 0; j < 8; ++j) {
        const int n = n0 + 16 * strip + j + 8 * lh;
        const int m = m0 + 32 * wcol + 16 * t + lr;
        av[t][j] = AB[(size_t)n * N_ + m];
      }
    // prefetch next adjacency tile while we compute on this one
    if (m0 + 64 < N_) {
      const int pr = threadIdx.x >> 2;
      const int pc = (threadIdx.x & 3) * 16;
      __builtin_prefetch(&AB[(size_t)(n0 + pr) * N_ + (m0 + 64) + pc], 0, 3);
    }

#pragma unroll 1
    for (int h = 0; h < H_; ++h) {
      // ---- GEMM1: S[64 x 64] = Ws[h] @ Xp_chunk^T (2 m-tiles per wave) ----
      v8f s[2] = {};
      {
        const __bf16* wsr = &Ws[h][16 * strip + lr][0];
#pragma unroll
        for (int kb = 0; kb < D_; kb += 32) {
          const v16bf a = cat8(*(const v8bf*)(wsr + kb + 16 * lh),
                               *(const v8bf*)(wsr + kb + 16 * lh + 8));
#pragma unroll
          for (int t = 0; t < 2; ++t) {
            const int m = m0 + 32 * wcol + 16 * t + lr;
            const v16bf bb = *(const v16bf*)(XpB + (size_t)m * D_ + kb + 16 * lh);
            s[t] = wmma_bf16(a, bb, s[t]);
          }
        }
      }
      // ---- elementwise: Am = tanh(A * S), redistributed via LDS ----
#pragma unroll
      for (int t = 0; t < 2; ++t)
#pragma unroll
        for (int j = 0; j < 8; ++j) {
          const float sv = fast_tanh(av[t][j] * s[t][j]);
          Am[16 * strip + j + 8 * lh][32 * wcol + 16 * t + lr] = f2bf(sv);
        }
      __syncthreads();
      // ---- GEMM2: hacc += Am @ Xp_chunk (B from XpT, 4 d-tiles/wave) ----
      {
        const __bf16* amr = &Am[16 * strip + lr][0];
#pragma unroll
        for (int kb = 0; kb < 64; kb += 32) {
          const v16bf a = cat8(*(const v8bf*)(amr + kb + 16 * lh),
                               *(const v8bf*)(amr + kb + 16 * lh + 8));
#pragma unroll
          for (int t = 0; t < 4; ++t) {
            const int d = 64 * wcol + 16 * t + lr;
            const v16bf bb =
                *(const v16bf*)(XpTB + (size_t)d * N_ + m0 + kb + 16 * lh);
            hacc[t] = wmma_bf16(a, bb, hacc[t]);
          }
        }
      }
      __syncthreads();   // Am is rewritten by the next head / next m-chunk
    }
  }

  // ---- epilogue: out = relu(relu(mean_h) + X) ----
#pragma unroll
  for (int t = 0; t < 4; ++t)
#pragma unroll
    for (int j = 0; j < 8; ++j) {
      const int n = n0 + 16 * strip + j + 8 * lh;
      const int d = 64 * wcol + 16 * t + lr;
      float v = hacc[t][j] * 0.25f;             // mean over H=4 heads
      v = fmaxf(v, 0.f);
      v += X[((size_t)b * N_ + n) * F_ + d];    // F == D skip connection
      v = fmaxf(v, 0.f);
      out[((size_t)b * N_ + n) * D_ + d] = v;
    }
}

// ---------------------------------------------------------------------------
extern "C" void kernel_launch(void* const* d_in, const int* in_sizes, int n_in,
                              void* d_out, int out_size, void* d_ws,
                              size_t ws_size, hipStream_t stream) {
  const float* X      = (const float*)d_in[0];   // [B,N,F]
  const float* A      = (const float*)d_in[1];   // [B,N,N]
  const float* kernel = (const float*)d_in[2];   // [F,D]
  const float* bias   = (const float*)d_in[3];   // [D]
  const float* attn   = (const float*)d_in[4];   // [H,D,D]
  float* out = (float*)d_out;

  char* ws = (char*)d_ws;
  __bf16* Xp    = (__bf16*)ws;                                    // 2 MB
  __bf16* XpT   = (__bf16*)(ws + (size_t)B_ * N_ * D_ * 2);       // 2 MB
  __bf16* attnT = (__bf16*)(ws + (size_t)B_ * N_ * D_ * 4);       // 128 KB

  k_prep<<<B_ * N_, 128, 0, stream>>>(X, kernel, bias, Xp, XpT);
  k_attnT<<<(H_ * D_ * D_) / 256, 256, 0, stream>>>(attn, attnT);
  k_main<<<B_ * (N_ / 64), 256, 0, stream>>>(X, A, Xp, XpT, attnT, out);
}
